// ArcLayer_38774964748998
// MI455X (gfx1250) — matrix operand order repaired and checked by reading, not compiled, not run on hardware
//
#include <hip/hip_runtime.h>
#include <hip/hip_bf16.h>

// ---------------------------------------------------------------------------
// ArcFace head on gfx1250 (MI455X), wave32.
//   xn  = l2norm_rows(x)   (f16)  [1024 x 512]
//   Wnt = l2norm_cols(W)^T (f16)  [100000 x 512]  (K-contiguous per class)
//   cos = xn @ Wn  via v_wmma_f32_16x16x32_f16, double-buffered LDS staged
//         with GLOBAL_LOAD_ASYNC_TO_LDS_B128 (ASYNCcnt), -> d_out as scratch
//   rowsum[b] = sum_c exp(S*cos[b,c])  (deterministic 2-level reduction)
//   out = e_phi / (e_phi + rowsum - e_cos), phi = c*cosM - sqrt(1-c^2)*sinM
// ---------------------------------------------------------------------------

typedef __attribute__((ext_vector_type(16))) _Float16 v16h;
typedef __attribute__((ext_vector_type(8)))  _Float16 v8h;
typedef __attribute__((ext_vector_type(8)))  float    v8f;
typedef __attribute__((ext_vector_type(4)))  float    v4f;

#define BATCH 1024
#define DDIM  512
#define NCLS  100000

#define BM 128
#define BN 128
#define BK 32
#define LDT 40            // padded LDS row stride (halves): 80B -> conflict-free b128
#define NSTEPS (DDIM / BK)            // 16
#define NBLK ((NCLS + BN - 1) / BN)   // 782

#define S_SCALE 30.0f
#define COS_M   0.98006657784124163f  // cos(0.2)
#define SIN_M   0.19866933079506122f  // sin(0.2)
#define CLIP_EPS 1e-7f

static __device__ inline v16h cat8(v8h lo, v8h hi) {
  return __builtin_shufflevector(lo, hi, 0,1,2,3,4,5,6,7,8,9,10,11,12,13,14,15);
}

// 16B global -> LDS async copy (per-lane), tracked by ASYNCcnt.
static __device__ inline void async_copy16(unsigned lds_off, const void* gaddr) {
  asm volatile("global_load_async_to_lds_b128 %0, %1, off"
               :: "v"(lds_off), "v"((unsigned long long)(uintptr_t)gaddr)
               : "memory");
}
#define WAIT_ASYNC(n) asm volatile("s_wait_asynccnt " #n ::: "memory")

static __device__ inline unsigned lds_off32(const void* p) {
  return (unsigned)(uintptr_t)p;   // generic shared ptr: LDS offset = addr[31:0]
}

// ---------------- kernel 1: normalize rows of x -> f16 ----------------------
__global__ __launch_bounds__(256) void xnorm_kernel(const float* __restrict__ x,
                                                    _Float16* __restrict__ xn) {
  const int row = blockIdx.x;
  const int tid = threadIdx.x;
  const float v0 = x[row * DDIM + tid];
  const float v1 = x[row * DDIM + tid + 256];
  __shared__ float red[256];
  red[tid] = v0 * v0 + v1 * v1;
  __syncthreads();
  for (int s = 128; s > 0; s >>= 1) {
    if (tid < s) red[tid] += red[tid + s];
    __syncthreads();
  }
  const float rn = rsqrtf(fmaxf(red[0], 1e-10f));
  xn[row * DDIM + tid]       = (_Float16)(v0 * rn);
  xn[row * DDIM + tid + 256] = (_Float16)(v1 * rn);
}

// -------- kernel 2: normalize cols of W -> f16, transposed [NCLS][DDIM] -----
__global__ __launch_bounds__(256) void wnorm_kernel(const float* __restrict__ W,
                                                    _Float16* __restrict__ Wnt) {
  const int c = blockIdx.x * 256 + threadIdx.x;
  if (c >= NCLS) return;
  float ss = 0.f;
  for (int k = 0; k < DDIM; ++k) {
    const float w = W[(size_t)k * NCLS + c];
    ss += w * w;
  }
  const float rn = rsqrtf(fmaxf(ss, 1e-10f));
  for (int k0 = 0; k0 < DDIM; k0 += 8) {
    v8h pack;
    #pragma unroll
    for (int j = 0; j < 8; ++j)
      pack[j] = (_Float16)(W[(size_t)(k0 + j) * NCLS + c] * rn);
    *(v8h*)(Wnt + (size_t)c * DDIM + k0) = pack;   // 16B store, K-contiguous
  }
}

// ---------------- kernel 3: WMMA GEMM + exp row partial sums ----------------
// grid = (8, 782) [m fastest -> Wnt slab reused from L2 across all 8 m-blocks]
// block = 256 (8 waves). Block tile 128x128, K-step 32, LDS double-buffered
// via async-to-LDS. Wave grid 2(m) x 4(n): 4x2 wmma accumulators per wave.
__global__ __launch_bounds__(256) void gemm_cos_kernel(
    const _Float16* __restrict__ xn, const _Float16* __restrict__ Wnt,
    float* __restrict__ cosout, float* __restrict__ rowpart) {
  __shared__ _Float16 At[2][BM * LDT];  // [row][k], K contiguous
  __shared__ _Float16 Bt[2][BN * LDT];  // [col][k], K contiguous
  __shared__ float    rp[4 * BM];

  const int tid   = threadIdx.x;
  const int lane  = tid & 31;
  const int wave  = tid >> 5;
  const int wm    = wave & 1;
  const int wn    = wave >> 1;
  const int mBase = blockIdx.x * BM;
  const int nBase = blockIdx.y * BN;

  const int rsel = lane & 15;
  const int hsel = lane >> 4;

  // staging geometry: 512 16B-chunks per tile, 2 per thread
  const int row0 = tid >> 2;             // 0..63
  const int row1 = row0 + 64;            // 64..127
  const int kc   = (tid & 3) * 8;        // 0,8,16,24
  // clamp OOB tail columns to a valid source; results discarded in epilogue
  const int bcol0 = min(nBase + row0, NCLS - 1);
  const int bcol1 = min(nBase + row1, NCLS - 1);

  auto stage = [&](int kk, int buf) {
    async_copy16(lds_off32(&At[buf][row0 * LDT + kc]),
                 xn + (size_t)(mBase + row0) * DDIM + kk + kc);
    async_copy16(lds_off32(&At[buf][row1 * LDT + kc]),
                 xn + (size_t)(mBase + row1) * DDIM + kk + kc);
    async_copy16(lds_off32(&Bt[buf][row0 * LDT + kc]),
                 Wnt + (size_t)bcol0 * DDIM + kk + kc);
    async_copy16(lds_off32(&Bt[buf][row1 * LDT + kc]),
                 Wnt + (size_t)bcol1 * DDIM + kk + kc);
  };

  v8f acc[4][2] = {};

  stage(0, 0);
  for (int i = 0; i < NSTEPS; ++i) {
    const int buf = i & 1;
    if (i + 1 < NSTEPS) {
      stage((i + 1) * BK, buf ^ 1);
      WAIT_ASYNC(4);         // in-order completion: stage i's 4 copies done
    } else {
      WAIT_ASYNC(0);
    }
    __syncthreads();

    // A 16x32 frag: lane row=lane%16, K base=(lane/16)*8; v0-3:K base..+7,
    // v4-7:K base+16..+23 -> two 16B LDS reads.
    v16h af[4];
    #pragma unroll
    for (int mi = 0; mi < 4; ++mi) {
      const _Float16* ab = &At[buf][(wm * 64 + mi * 16 + rsel) * LDT + hsel * 8];
      af[mi] = cat8(*(const v8h*)ab, *(const v8h*)(ab + 16));
    }
    // B 32x16 frag: lane col=lane%16, K base=(lane/16)*8; v0-3:K base..+7, v4-7:K base+16..+23.
    v16h bf[2];
    #pragma unroll
    for (int ni = 0; ni < 2; ++ni) {
      const _Float16* bb = &Bt[buf][(wn * 32 + ni * 16 + rsel) * LDT + hsel * 8];
      bf[ni] = cat8(*(const v8h*)bb, *(const v8h*)(bb + 16));
    }
    #pragma unroll
    for (int mi = 0; mi < 4; ++mi)
      #pragma unroll
      for (int ni = 0; ni < 2; ++ni)
        acc[mi][ni] = __builtin_amdgcn_wmma_f32_16x16x32_f16(
            false, af[mi], false, bf[ni], (short)0, acc[mi][ni], false, false);
    __syncthreads();
  }

  // --- epilogue: NT-store cos, deterministic exp-rowsum partials ---
  // C/D layout: VGPR j -> M = j + 8*(lane/16); N = lane%16.
  #pragma unroll
  for (int mi = 0; mi < 4; ++mi) {
    #pragma unroll
    for (int j = 0; j < 8; ++j) {
      const int row_t = wm * 64 + mi * 16 + j + 8 * hsel;
      const int grow  = mBase + row_t;
      float esum = 0.f;
      #pragma unroll
      for (int ni = 0; ni < 2; ++ni) {
        const float c = acc[mi][ni][j];
        const int gcol = nBase + wn * 32 + ni * 16 + rsel;
        float e = 0.f;
        if (gcol < NCLS) {
          __builtin_nontemporal_store(c, cosout + (size_t)grow * NCLS + gcol);
          e = __expf(S_SCALE * c);
        }
        #pragma unroll
        for (int off = 1; off < 16; off <<= 1) e += __shfl_xor(e, off, 32);
        esum += e;
      }
      if (rsel == 0) rp[wn * BM + row_t] = esum;  // lanes 0 and 16
    }
  }
  __syncthreads();
  if (tid < BM) {
    const float s = rp[tid] + rp[BM + tid] + rp[2 * BM + tid] + rp[3 * BM + tid];
    rowpart[(size_t)(mBase + tid) * NBLK + blockIdx.y] = s;
  }
}

// ---------------- kernel 4: fixed-order row-sum reduction -------------------
__global__ __launch_bounds__(256) void rowreduce_kernel(const float* __restrict__ rowpart,
                                                        float* __restrict__ rowsum) {
  const int r = blockIdx.x * 256 + threadIdx.x;
  if (r >= BATCH) return;
  float s = 0.f;
  for (int nb = 0; nb < NBLK; ++nb) s += rowpart[(size_t)r * NBLK + nb];
  rowsum[r] = s;
}

// ---------------- kernel 5: in-place ArcFace epilogue -----------------------
__global__ __launch_bounds__(256) void finalize_kernel(float* __restrict__ out_io,
                                                       const float* __restrict__ rowsum) {
  const size_t i4   = (size_t)blockIdx.x * 256 + threadIdx.x;
  const size_t base = i4 * 4;                      // NCLS % 4 == 0: no row straddle
  const int row = (int)(base / NCLS);
  const float rs = rowsum[row];
  v4f c4 = __builtin_nontemporal_load((const v4f*)(out_io + base));
  v4f o4;
  #pragma unroll
  for (int i = 0; i < 4; ++i) {
    const float c  = c4[i];
    const float cc = fminf(fmaxf(c, -1.0f + CLIP_EPS), 1.0f - CLIP_EPS);
    const float phi = cc * COS_M - sqrtf(fmaxf(1.0f - cc * cc, 0.0f)) * SIN_M;
    const float e_phi = __expf(S_SCALE * phi);
    const float e_cos = __expf(S_SCALE * c);
    o4[i] = e_phi / (e_phi + (rs - e_cos));
  }
  __builtin_nontemporal_store(o4, (v4f*)(out_io + base));
}

// ---------------------------------------------------------------------------
extern "C" void kernel_launch(void* const* d_in, const int* in_sizes, int n_in,
                              void* d_out, int out_size, void* d_ws, size_t ws_size,
                              hipStream_t stream) {
  const float* x = (const float*)d_in[0];   // [1024, 512]
  const float* W = (const float*)d_in[1];   // [512, 100000]
  float* out = (float*)d_out;               // [1024, 100000]

  // workspace: xn f16 | Wnt f16 (transposed) | rowpart f32 | rowsum f32
  char* ws = (char*)d_ws;
  const size_t XN_BYTES = (size_t)BATCH * DDIM * sizeof(_Float16);   // 1.0 MB
  const size_t WN_BYTES = (size_t)NCLS * DDIM * sizeof(_Float16);    // 102.4 MB
  const size_t RP_BYTES = (size_t)BATCH * NBLK * sizeof(float);      // 3.2 MB
  _Float16* xn      = (_Float16*)ws;
  _Float16* Wnt     = (_Float16*)(ws + XN_BYTES);
  float*    rowpart = (float*)(ws + XN_BYTES + WN_BYTES);
  float*    rowsum  = (float*)(ws + XN_BYTES + WN_BYTES + RP_BYTES);

  xnorm_kernel<<<BATCH, 256, 0, stream>>>(x, xn);
  wnorm_kernel<<<(NCLS + 255) / 256, 256, 0, stream>>>(W, Wnt);
  gemm_cos_kernel<<<dim3(8, NBLK), 256, 0, stream>>>(xn, Wnt, out, rowpart);
  rowreduce_kernel<<<4, 256, 0, stream>>>(rowpart, rowsum);
  finalize_kernel<<<NCLS, 256, 0, stream>>>(out, rowsum);
}
